// HGTLayer_55370718380398
// MI455X (gfx1250) — compile-verified, matrix operand order checked by reading, not run on hardware
//
#include <hip/hip_runtime.h>
#include <hip/hip_bf16.h>

#define N_NODES 40000
#define N_EDGES 640000

typedef _Float16 half16 __attribute__((ext_vector_type(16)));
typedef float    float8 __attribute__((ext_vector_type(8)));

// ---------------------------------------------------------------------------
// Packed B-fragment layout for WMMA f16 16x16x32:
//   packed[coltile(8)][kchunk(4)][lane(32)][elem(16)]  (halves)
// lane = (n%16) + 16*khalf ; elem i -> k_in_chunk = 16*(i/8) + 8*khalf + i%8
// Each lane's fragment is 32 contiguous bytes -> b128-pair loads.
// ---------------------------------------------------------------------------
__device__ __forceinline__ int pack_idx(int k, int c) {
    int coltile = c >> 4, n16 = c & 15;
    int chunk = k >> 5, kk = k & 31;
    int khalf = (kk >> 3) & 1;
    int i = ((kk >> 4) << 3) + (kk & 7);
    int lane = n16 + (khalf << 4);
    return coltile * 2048 + chunk * 512 + lane * 16 + i;
}

// Fuse per-head 16x16 relation transform into projection weight, emit packed
// f16: Wout_packed[(k, h*16+e)] = sum_d W[k, h*16+d] * R[h, d, e]
__global__ void fuse_w_kernel(const float* __restrict__ W,
                              const float* __restrict__ R,
                              _Float16* __restrict__ Wout) {
    int idx = blockIdx.x * blockDim.x + threadIdx.x;   // 0..16383
    int k = idx >> 7;
    int c = idx & 127;
    int h = c >> 4;
    int e = c & 15;
    float s = 0.f;
#pragma unroll
    for (int d = 0; d < 16; ++d)
        s += W[k * 128 + h * 16 + d] * R[h * 256 + d * 16 + e];
    Wout[pack_idx(k, c)] = (_Float16)s;
}

__global__ void fuse_b_kernel(const float* __restrict__ b,
                              const float* __restrict__ R,
                              float* __restrict__ bout) {
    int c = threadIdx.x;          // 0..127
    int h = c >> 4;
    int e = c & 15;
    float s = 0.f;
#pragma unroll
    for (int d = 0; d < 16; ++d)
        s += b[h * 16 + d] * R[h * 256 + d * 16 + e];
    bout[c] = s;
}

// straight f32 -> packed f16 conversion
__global__ void convert_w_kernel(const float* __restrict__ W,
                                 _Float16* __restrict__ Wout) {
    int idx = blockIdx.x * blockDim.x + threadIdx.x;   // 0..16383
    int k = idx >> 7;
    int c = idx & 127;
    Wout[pack_idx(k, c)] = (_Float16)W[idx];
}

__global__ void zero_f_kernel(float* __restrict__ p, int n) {
    int i = blockIdx.x * blockDim.x + threadIdx.x;
    if (i < n) p[i] = 0.f;
}

// ---------------------------------------------------------------------------
// WMMA GEMM: Y[nrows,128] = f16(X[nrows,128]) @ W(packed f16 128x128) + bias
//            (+ optional residual).
// Block = 256 threads = 8 waves (one per 16-col tile); each wave keeps its
// four B fragments resident in VGPRs and sweeps 4 row tiles (64 rows/block).
// EXEC stays all-ones (nrows % 64 == 0, no early exits) as WMMA requires.
// ---------------------------------------------------------------------------
__global__ __launch_bounds__(256)
void gemm_wmma_kernel(const float* __restrict__ X,
                      const _Float16* __restrict__ Wpk,
                      const float* __restrict__ bias,
                      const float* __restrict__ resid,
                      float* __restrict__ Y) {
    const int lane  = threadIdx.x & 31;
    const int wave  = threadIdx.x >> 5;        // col tile 0..7
    const int n16   = lane & 15;
    const int khalf = lane >> 4;
    const int ncol  = wave * 16 + n16;

    // B fragments: 32 contiguous bytes per lane per K-chunk
    half16 bfrag[4];
    const half16* bp = (const half16*)(Wpk + (size_t)wave * 2048);
#pragma unroll
    for (int kc = 0; kc < 4; ++kc)
        bfrag[kc] = bp[kc * 32 + lane];

    const float bval = bias[ncol];

#pragma unroll
    for (int rt = 0; rt < 4; ++rt) {
        const int mbase = blockIdx.x * 64 + rt * 16;
        const int m = mbase + n16;             // A row held by this lane
        float8 c = {};
#pragma unroll
        for (int kc = 0; kc < 4; ++kc) {
            half16 a;
            const float* xp = X + (size_t)m * 128 + kc * 32 + khalf * 8;
#pragma unroll
            for (int i = 0; i < 8; ++i) {      // two contiguous 8-float runs
                a[i]     = (_Float16)xp[i];
                a[i + 8] = (_Float16)xp[16 + i];
            }
            c = __builtin_amdgcn_wmma_f32_16x16x32_f16(
                    /*neg_a=*/false, a, /*neg_b=*/false, bfrag[kc],
                    /*c_mod=*/(short)0, c, /*reuse_a=*/false, /*reuse_b=*/false);
        }
#pragma unroll
        for (int v = 0; v < 8; ++v) {
            // f32 C/D layout: VGPR v -> M = v + 8*khalf, N = lane%16
            int mo = mbase + v + (khalf << 3);
            float val = c[v] + bval;
            if (resid) val += resid[(size_t)mo * 128 + ncol];
            Y[(size_t)mo * 128 + ncol] = val;
        }
    }
}

// ---------------------------------------------------------------------------
// Edge phase: segment softmax (3 passes) + message scatter
// ---------------------------------------------------------------------------
__device__ __forceinline__ unsigned f2key(float f) {
    unsigned u = __float_as_uint(f);
    return (u & 0x80000000u) ? ~u : (u | 0x80000000u);
}
__device__ __forceinline__ float key2f(unsigned k) {
    return (k & 0x80000000u) ? __uint_as_float(k ^ 0x80000000u)
                             : __uint_as_float(~k);
}

// one thread per (edge, head): score = <q[dst,h,:], k2[src,h,:]> * pri/4
__global__ __launch_bounds__(256)
void score_kernel(const float* __restrict__ q, const float* __restrict__ k2,
                  const int* __restrict__ src, const int* __restrict__ dst,
                  const float* __restrict__ pri,
                  float* __restrict__ SC, unsigned* __restrict__ MX) {
    int t = blockIdx.x * blockDim.x + threadIdx.x;
    if (t >= N_EDGES * 8) return;
    int e = t >> 3, h = t & 7;
    int s = src[e], d = dst[e];
    const float4* qv = (const float4*)(q  + (size_t)d * 128 + h * 16);
    const float4* kv = (const float4*)(k2 + (size_t)s * 128 + h * 16);
    float acc = 0.f;
#pragma unroll
    for (int j = 0; j < 4; ++j) {
        float4 a = qv[j], b = kv[j];
        acc += a.x * b.x + a.y * b.y + a.z * b.z + a.w * b.w;
    }
    float sc = acc * pri[h] * 0.25f;   // 1/sqrt(DK=16)
    SC[t] = sc;
    atomicMax(&MX[d * 8 + h], f2key(sc));
}

__global__ __launch_bounds__(256)
void expsum_kernel(const int* __restrict__ dst, const unsigned* __restrict__ MX,
                   float* __restrict__ SC, float* __restrict__ Z) {
    int t = blockIdx.x * blockDim.x + threadIdx.x;
    if (t >= N_EDGES * 8) return;
    int e = t >> 3, h = t & 7;
    int d = dst[e];
    float m  = key2f(MX[d * 8 + h]);
    float es = __expf(SC[t] - m);
    SC[t] = es;
    atomicAdd(&Z[d * 8 + h], es);
}

// one wave per edge, each lane scatters 4 of the 128 message elements
__global__ __launch_bounds__(256)
void scatter_kernel(const int* __restrict__ src, const int* __restrict__ dst,
                    const float* __restrict__ SC, const float* __restrict__ Z,
                    const float* __restrict__ v2, float* __restrict__ agg) {
    int t = blockIdx.x * blockDim.x + threadIdx.x;
    int e = t >> 5;
    if (e >= N_EDGES) return;
    int l = t & 31;
    int s = src[e], d = dst[e];
#pragma unroll
    for (int j4 = 0; j4 < 4; ++j4) {
        int j = l + j4 * 32;
        int h = j >> 4;
        float attn = SC[e * 8 + h] / Z[d * 8 + h];
        atomicAdd(&agg[(size_t)d * 128 + j], attn * v2[(size_t)s * 128 + j]);
    }
}

// ---------------------------------------------------------------------------
// In-place LayerNorm over 128 features; one wave32 per row, shfl reductions
// ---------------------------------------------------------------------------
__global__ __launch_bounds__(256)
void layernorm_kernel(float* __restrict__ Y, const float* __restrict__ g,
                      const float* __restrict__ b) {
    int t = blockIdx.x * blockDim.x + threadIdx.x;
    int row = t >> 5;
    if (row >= N_NODES) return;
    int l = t & 31;
    float x[4];
    float s = 0.f, s2 = 0.f;
#pragma unroll
    for (int j = 0; j < 4; ++j) {
        x[j] = Y[(size_t)row * 128 + l + j * 32];
        s  += x[j];
        s2 += x[j] * x[j];
    }
#pragma unroll
    for (int off = 16; off > 0; off >>= 1) {
        s  += __shfl_xor(s,  off, 32);
        s2 += __shfl_xor(s2, off, 32);
    }
    float mu  = s * (1.f / 128.f);
    float var = s2 * (1.f / 128.f) - mu * mu;
    float inv = rsqrtf(var + 1e-5f);
#pragma unroll
    for (int j = 0; j < 4; ++j) {
        int cidx = l + j * 32;
        Y[(size_t)row * 128 + cidx] = (x[j] - mu) * inv * g[cidx] + b[cidx];
    }
}

// ---------------------------------------------------------------------------
// Launch.  Workspace layout (floats):
//   [0..8)*NF : qA qB k2_0 v2_0 k2_1 v2_1 aggA aggB      (NF = N*128)
//   8*NF      : SC (E*8 == NF floats, reused across relations; stream order
//               serializes)
//   9*NF      : MX0, Z0, MX1, Z1 (N*8 each)
//   then      : 8 packed-f16 weight matrices (16384 halves each), 4 biases
// Total ~190 MB.
// ---------------------------------------------------------------------------
extern "C" void kernel_launch(void* const* d_in, const int* in_sizes, int n_in,
                              void* d_out, int out_size, void* d_ws, size_t ws_size,
                              hipStream_t stream) {
    const float* h_A      = (const float*)d_in[0];
    const float* h_B      = (const float*)d_in[1];
    const int*   src0     = (const int*)  d_in[2];
    const int*   dst0     = (const int*)  d_in[3];
    const int*   src1     = (const int*)  d_in[4];
    const int*   dst1     = (const int*)  d_in[5];
    const float* k_w      = (const float*)d_in[6];
    const float* k_b      = (const float*)d_in[7];
    const float* q_w      = (const float*)d_in[8];
    const float* q_b      = (const float*)d_in[9];
    const float* v_w      = (const float*)d_in[10];
    const float* v_b      = (const float*)d_in[11];
    const float* a_w      = (const float*)d_in[12];
    const float* a_b      = (const float*)d_in[13];
    const float* ln_scale = (const float*)d_in[14];
    const float* ln_bias  = (const float*)d_in[15];
    const float* rel_pri  = (const float*)d_in[16];
    const float* rel_att  = (const float*)d_in[17];
    const float* rel_msg  = (const float*)d_in[18];

    float* ws = (float*)d_ws;
    const size_t NF = (size_t)N_NODES * 128;

    float* qA   = ws + 0 * NF;
    float* qB   = ws + 1 * NF;
    float* k20  = ws + 2 * NF;
    float* v20  = ws + 3 * NF;
    float* k21  = ws + 4 * NF;
    float* v21  = ws + 5 * NF;
    float* aggA = ws + 6 * NF;
    float* aggB = ws + 7 * NF;
    float* SC   = ws + 8 * NF;                    // E*8 floats
    float*    stats = ws + 9 * NF;                // MX0,Z0,MX1,Z1
    unsigned* MX0 = (unsigned*)(stats);
    float*    Z0  = stats + 1 * (size_t)N_NODES * 8;
    unsigned* MX1 = (unsigned*)(stats + 2 * (size_t)N_NODES * 8);
    float*    Z1  = stats + 3 * (size_t)N_NODES * 8;

    _Float16* WH   = (_Float16*)(stats + 4 * (size_t)N_NODES * 8);
    _Float16* W_q0  = WH + 0 * 16384;
    _Float16* W_q1  = WH + 1 * 16384;
    _Float16* W_k20 = WH + 2 * 16384;
    _Float16* W_v20 = WH + 3 * 16384;
    _Float16* W_k21 = WH + 4 * 16384;
    _Float16* W_v21 = WH + 5 * 16384;
    _Float16* W_a0  = WH + 6 * 16384;
    _Float16* W_a1  = WH + 7 * 16384;
    float* BB    = (float*)(WH + 8 * 16384);
    float* b_k20 = BB + 0;
    float* b_v20 = BB + 128;
    float* b_k21 = BB + 256;
    float* b_v21 = BB + 384;

    // ---- weight prep (fold per-head relation matrices into projections) ----
    fuse_w_kernel<<<64, 256, 0, stream>>>(k_w,         rel_att,        W_k20);
    fuse_w_kernel<<<64, 256, 0, stream>>>(v_w,         rel_msg,        W_v20);
    fuse_w_kernel<<<64, 256, 0, stream>>>(k_w + 16384, rel_att + 2048, W_k21);
    fuse_w_kernel<<<64, 256, 0, stream>>>(v_w + 16384, rel_msg + 2048, W_v21);
    convert_w_kernel<<<64, 256, 0, stream>>>(q_w,         W_q0);
    convert_w_kernel<<<64, 256, 0, stream>>>(q_w + 16384, W_q1);
    convert_w_kernel<<<64, 256, 0, stream>>>(a_w,         W_a0);
    convert_w_kernel<<<64, 256, 0, stream>>>(a_w + 16384, W_a1);
    fuse_b_kernel<<<1, 128, 0, stream>>>(k_b,       rel_att,        b_k20);
    fuse_b_kernel<<<1, 128, 0, stream>>>(v_b,       rel_msg,        b_v20);
    fuse_b_kernel<<<1, 128, 0, stream>>>(k_b + 128, rel_att + 2048, b_k21);
    fuse_b_kernel<<<1, 128, 0, stream>>>(v_b + 128, rel_msg + 2048, b_v21);

    // ---- init accumulators / softmax stats (ws is poisoned by harness) ----
    {
        int n = (int)(2 * NF);
        zero_f_kernel<<<(n + 255) / 256, 256, 0, stream>>>(aggA, n);
        int m = 4 * N_NODES * 8;
        zero_f_kernel<<<(m + 255) / 256, 256, 0, stream>>>(stats, m);
    }

    // ---- projection GEMMs (all WMMA); 64 rows per block ----
    const int GB = N_NODES / 64;  // 625 blocks
    gemm_wmma_kernel<<<GB, 256, 0, stream>>>(h_A, W_q0,  q_b,       nullptr, qA);
    gemm_wmma_kernel<<<GB, 256, 0, stream>>>(h_B, W_q1,  q_b + 128, nullptr, qB);
    gemm_wmma_kernel<<<GB, 256, 0, stream>>>(h_A, W_k20, b_k20,     nullptr, k20);
    gemm_wmma_kernel<<<GB, 256, 0, stream>>>(h_A, W_v20, b_v20,     nullptr, v20);
    gemm_wmma_kernel<<<GB, 256, 0, stream>>>(h_B, W_k21, b_k21,     nullptr, k21);
    gemm_wmma_kernel<<<GB, 256, 0, stream>>>(h_B, W_v21, b_v21,     nullptr, v21);

    // ---- relation 0: A -> B (dst are B nodes, query = qB) ----
    score_kernel  <<<20000, 256, 0, stream>>>(qB, k20, src0, dst0, rel_pri,     SC, MX0);
    expsum_kernel <<<20000, 256, 0, stream>>>(dst0, MX0, SC, Z0);
    scatter_kernel<<<80000, 256, 0, stream>>>(src0, dst0, SC, Z0, v20, aggB);

    // ---- relation 1: B -> A (dst are A nodes, query = qA) ----
    score_kernel  <<<20000, 256, 0, stream>>>(qA, k21, src1, dst1, rel_pri + 8, SC, MX1);
    expsum_kernel <<<20000, 256, 0, stream>>>(dst1, MX1, SC, Z1);
    scatter_kernel<<<80000, 256, 0, stream>>>(src1, dst1, SC, Z1, v21, aggA);

    // ---- output projection + residual (WMMA), then LayerNorm in-place ----
    float* outA = (float*)d_out;
    float* outB = outA + NF;
    gemm_wmma_kernel<<<GB, 256, 0, stream>>>(aggA, W_a0, a_b,       h_A, outA);
    gemm_wmma_kernel<<<GB, 256, 0, stream>>>(aggB, W_a1, a_b + 128, h_B, outB);
    layernorm_kernel<<<5000, 256, 0, stream>>>(outA, ln_scale,       ln_bias);
    layernorm_kernel<<<5000, 256, 0, stream>>>(outB, ln_scale + 128, ln_bias + 128);
}
